// MultiHeadAttention_84224308674935
// MI455X (gfx1250) — matrix operand order repaired and checked by reading, not compiled, not run on hardware
//
// MI455X (gfx1250) fused multi-head attention, bf16 WMMA (v_wmma_f32_16x16x32_bf16)
// Pipeline: cast/pack -> GEMM(Q|V) -> GEMM(K) -> flash-attention -> GEMM(Wp->concat) -> GEMM(Wo)
// GEMM: async-to-LDS double-buffered staging (ASYNCcnt), dynamic LDS (73728B).
// Attention: async-to-LDS single-buffer staging.
#include <hip/hip_runtime.h>

#define DEV static __device__ __forceinline__

typedef __attribute__((ext_vector_type(16))) __bf16 bfx16;
typedef __attribute__((ext_vector_type(8)))  __bf16 bfx8;
typedef __attribute__((ext_vector_type(8)))  float  fx8;
typedef __attribute__((ext_vector_type(4)))  int    v4i;

#define BB     4
#define SQL    2048
#define SKL    2048
#define INDIM  1024
#define DM     128
#define NH     16

#if __has_builtin(__builtin_amdgcn_global_load_async_to_lds_b128) && \
    __has_builtin(__builtin_amdgcn_s_wait_asynccnt)
#define HAVE_ASYNC_LDS 1
#else
#define HAVE_ASYNC_LDS 0
#endif

// ---------- helpers ----------
DEV unsigned short f2bf(float f) {
  union { float f; unsigned u; } x; x.f = f;
  unsigned r = x.u + 0x7FFFu + ((x.u >> 16) & 1u);
  return (unsigned short)(r >> 16);
}

DEV bfx16 ld_frag2(const __bf16* p0, const __bf16* p1) {
  bfx16 r;
  bfx8 a = *(const bfx8*)p0;
  bfx8 b = *(const bfx8*)p1;
#pragma unroll
  for (int i = 0; i < 8; ++i) { r[i] = a[i]; r[8 + i] = b[i]; }
  return r;
}

// A-operand 16x32 bf16 frag: lanes 0-15 hold K{0..7,16..23}, lanes 16-31 K{8..15,24..31}
DEV bfx16 ld_afrag(const __bf16* rowk, int lane) {
  int klo = (lane >= 16) ? 8 : 0;
  return ld_frag2(rowk + klo, rowk + klo + 16);
}
// B-operand 32x16 bf16 frag (stored as Bt rows = N, contiguous K): lanes 0-15 K{0..7,16..23}, lanes 16-31 K{8..15,24..31}
DEV bfx16 ld_bfrag(const __bf16* rowk, int lane) {
  int kb = (lane >= 16) ? 8 : 0;
  return ld_frag2(rowk + kb, rowk + kb + 16);
}

DEV fx8 wmma_bf16(bfx16 a, bfx16 b, fx8 c) {
  return __builtin_amdgcn_wmma_f32_16x16x32_bf16(false, a, false, b, (short)0, c, false, false);
}

#if HAVE_ASYNC_LDS
DEV void async_cp16(const unsigned short* g, unsigned short* l) {
  __builtin_amdgcn_global_load_async_to_lds_b128(
      (__attribute__((address_space(1))) v4i*)g,
      (__attribute__((address_space(3))) v4i*)l, 0, 0);
}
#endif

// ---------- cast / pack kernels ----------
__global__ void k_cast(const float* __restrict__ src, unsigned short* __restrict__ dst, int n) {
  int i = (blockIdx.x * 256 + threadIdx.x) * 4;
  if (i >= n) return;
  float4 v = *(const float4*)(src + i);
  unsigned long long p = (unsigned long long)f2bf(v.x)
                       | ((unsigned long long)f2bf(v.y) << 16)
                       | ((unsigned long long)f2bf(v.z) << 32)
                       | ((unsigned long long)f2bf(v.w) << 48);
  *(unsigned long long*)(dst + i) = p;
}

// WqvT[4096][1024]: rows 0..2047 = Wq^T (n=h*128+d), rows 2048..4095 = Wv^T
__global__ void k_pack_wqv(const float* __restrict__ Wq, const float* __restrict__ Wv,
                           unsigned short* __restrict__ dst) {
  int idx = blockIdx.x * 256 + threadIdx.x;   // 0 .. 4M-1
  int n = idx >> 10, k = idx & 1023;
  float v;
  if (n < 2048) { int h = n >> 7, d = n & 127; v = Wq[((size_t)(h << 10) + k) * DM + d]; }
  else { int nn = n - 2048; int h = nn >> 7, d = nn & 127; v = Wv[((size_t)(h << 10) + k) * DM + d]; }
  dst[idx] = f2bf(v);
}

__global__ void k_pack_wk(const float* __restrict__ Wk, unsigned short* __restrict__ dst) {
  int idx = blockIdx.x * 256 + threadIdx.x;   // 0 .. 2M-1
  int n = idx >> 10, k = idx & 1023;
  int h = n >> 7, d = n & 127;
  dst[idx] = f2bf(Wk[((size_t)(h << 10) + k) * DM + d]);
}

// WpT[h][e][d] = Wp[h][d][e]
__global__ void k_pack_wp(const float* __restrict__ Wp, unsigned short* __restrict__ dst) {
  int idx = blockIdx.x * 256 + threadIdx.x;   // 0 .. 256K-1
  int h = idx >> 14, e = (idx >> 7) & 127, d = idx & 127;
  dst[idx] = f2bf(Wp[((size_t)(h * DM) + d) * DM + e]);
}

// WoT[e][hd] = Wo[hd][e]
__global__ void k_pack_wo(const float* __restrict__ Wo, unsigned short* __restrict__ dst) {
  int idx = blockIdx.x * 256 + threadIdx.x;   // 0 .. 256K-1
  int e = idx >> 11, hd = idx & 2047;
  dst[idx] = f2bf(Wo[(size_t)hd * DM + e]);
}

// ---------- generic bf16 WMMA GEMM: C[128x128 tile] = A[M,K] * Bt[N,K]^T ----------
// Dynamic LDS layout (ushort units): buf p: sA = p*18432, sB = p*18432 + 9216 (rows padded to 72)
// mode 0: QV proj  (N=4096: cols<2048 -> Qb[H][B][SQ][D]+bq, else -> Vt[H][B][D][SK]+bv)
// mode 1: K  proj  (N=2048 -> Kb[H][B][SK][D]+bk)
// mode 2: Wp proj  (A=CTX[H*B*SQ][128], per-head Bt; out -> CAT[B*SQ][H*128]+bp)
// mode 3: out proj (A=CAT, Bt=WoT; out -> float d_out[B*SQ][128]+bo)
__global__ __launch_bounds__(256)
void k_gemm(const unsigned short* __restrict__ A, int lda,
            const unsigned short* __restrict__ Bt, int ldb,
            int K, int mode,
            const float* __restrict__ bias0, const float* __restrict__ bias1,
            unsigned short* __restrict__ out0, unsigned short* __restrict__ out1,
            float* __restrict__ outf) {
  extern __shared__ __align__(16) unsigned short smem[];
  const int tid = threadIdx.x, lane = tid & 31, w = tid >> 5;
  const int blockM = blockIdx.y * 128, blockN = blockIdx.x * 128;
  if (mode == 2) Bt += (size_t)(blockM >> 13) * (DM * DM);   // per-head Wp

  const int mw = (w & 3) * 32, nw = (w >> 2) * 64;
  const int srow = tid >> 3, scol = (tid & 7) * 8;           // staging coords (8x16B per row)
  fx8 acc[2][4];
#pragma unroll
  for (int ms = 0; ms < 2; ++ms)
#pragma unroll
    for (int ns = 0; ns < 4; ++ns) acc[ms][ns] = {};

  auto do_compute = [&](const unsigned short* cA, const unsigned short* cB) {
#pragma unroll
    for (int mk = 0; mk < 2; ++mk) {
      bfx16 af[2], bfm[4];
#pragma unroll
      for (int ms = 0; ms < 2; ++ms)
        af[ms] = ld_afrag((const __bf16*)&cA[(mw + ms * 16 + (lane & 15)) * 72 + mk * 32], lane);
#pragma unroll
      for (int ns = 0; ns < 4; ++ns)
        bfm[ns] = ld_bfrag((const __bf16*)&cB[(nw + ns * 16 + (lane & 15)) * 72 + mk * 32], lane);
#pragma unroll
      for (int ms = 0; ms < 2; ++ms)
#pragma unroll
        for (int ns = 0; ns < 4; ++ns)
          acc[ms][ns] = wmma_bf16(af[ms], bfm[ns], acc[ms][ns]);
    }
  };

#if HAVE_ASYNC_LDS
  auto stage_async = [&](int kkx, int p) {                   // 8 async b128 per thread
    unsigned short* dA = smem + p * 18432;
    unsigned short* dB = dA + 9216;
#pragma unroll
    for (int it = 0; it < 4; ++it) {
      int row = srow + it * 32;
      async_cp16(&A[(size_t)(blockM + row) * lda + kkx + scol], &dA[row * 72 + scol]);
      async_cp16(&Bt[(size_t)(blockN + row) * ldb + kkx + scol], &dB[row * 72 + scol]);
    }
  };
  const int nstage = K >> 6;
  stage_async(0, 0);                                         // prologue: stage 0 -> buf 0
  for (int i = 0; i < nstage; ++i) {
    int p = i & 1;
    if (i + 1 < nstage) {                                    // overlap next DMA with compute
      stage_async((i + 1) << 6, 1 - p);
      __builtin_amdgcn_s_wait_asynccnt(8);                   // stage i landed (in-order)
    } else {
      __builtin_amdgcn_s_wait_asynccnt(0);
    }
    __syncthreads();                                         // all waves see stage i
    do_compute(smem + p * 18432, smem + p * 18432 + 9216);
    __syncthreads();                                         // readers done before reuse
  }
#else
  unsigned short* sA = smem;
  unsigned short* sB = smem + 9216;
  for (int kk = 0; kk < K; kk += 64) {
    __syncthreads();
#pragma unroll
    for (int it = 0; it < 4; ++it) {
      int row = srow + it * 32;
      *(int4*)&sA[row * 72 + scol] = *(const int4*)&A[(size_t)(blockM + row) * lda + kk + scol];
      *(int4*)&sB[row * 72 + scol] = *(const int4*)&Bt[(size_t)(blockN + row) * ldb + kk + scol];
    }
    __syncthreads();
    do_compute(sA, sB);
  }
#endif

  const int mhi = (lane >= 16) ? 8 : 0;
#pragma unroll
  for (int ms = 0; ms < 2; ++ms)
#pragma unroll
    for (int ns = 0; ns < 4; ++ns) {
      int gn = blockN + nw + ns * 16 + (lane & 15);
#pragma unroll
      for (int r = 0; r < 8; ++r) {
        int gm = blockM + mw + ms * 16 + r + mhi;
        float v = acc[ms][ns][r];
        if (mode == 0) {
          int b = gm >> 11, s = gm & 2047;
          if (gn < 2048) {
            int h = gn >> 7, d = gn & 127;
            out0[((size_t)(h * BB + b) * SQL + s) * DM + d] = f2bf(v + bias0[gn]);
          } else {
            int nn = gn - 2048; int h = nn >> 7, d = nn & 127;
            out1[(((size_t)(h * BB + b)) * DM + d) * SKL + s] = f2bf(v + bias1[nn]);
          }
        } else if (mode == 1) {
          int b = gm >> 11, s = gm & 2047;
          int h = gn >> 7, d = gn & 127;
          out0[((size_t)(h * BB + b) * SKL + s) * DM + d] = f2bf(v + bias0[gn]);
        } else if (mode == 2) {
          int h = gm >> 13, rem = gm & 8191;
          out0[(size_t)rem * 2048 + h * DM + gn] = f2bf(v + bias0[h * DM + gn]);
        } else {
          outf[(size_t)gm * DM + gn] = v + bias0[gn];
        }
      }
    }
}

// ---------- flash attention: 128 q-rows/block, 64-key tiles, online softmax ----------
__global__ __launch_bounds__(256)
void k_attn(const unsigned short* __restrict__ Qb, const unsigned short* __restrict__ Kb,
            const unsigned short* __restrict__ Vt, unsigned short* __restrict__ CTX) {
  __shared__ __align__(16) unsigned short sK[64 * 136];     // 64 keys x 128 d
  __shared__ __align__(16) unsigned short sV[128 * 72];     // V^T: 128 d x 64 keys
  __shared__ __align__(16) unsigned short sP[8 * 16 * 72];  // per-wave P 16x64
  const int tid = threadIdx.x, lane = tid & 31, w = tid >> 5;
  const int h = blockIdx.z, b = blockIdx.y, q0 = blockIdx.x * 128;
  const size_t hb = (size_t)h * BB + b;
  const unsigned short* Qp = Qb + (hb * SQL + q0) * DM;
  const unsigned short* Kp = Kb + hb * SKL * DM;
  const unsigned short* Vp = Vt + hb * (size_t)DM * SKL;
  const float scale = 0.08838834764831845f;  // 1/sqrt(128)

  bfx16 qf[4];                                              // Q frags held in regs
  {
    const __bf16* qrow = (const __bf16*)(Qp + (size_t)(w * 16 + (lane & 15)) * DM);
#pragma unroll
    for (int ks = 0; ks < 4; ++ks) qf[ks] = ld_afrag(qrow + ks * 32, lane);
  }
  fx8 cacc[8];
#pragma unroll
  for (int i = 0; i < 8; ++i) cacc[i] = {};
  float m[8], l[8];
#pragma unroll
  for (int r = 0; r < 8; ++r) { m[r] = -1e30f; l[r] = 0.f; }
  unsigned short* myP = &sP[w * 16 * 72];

  const int krow = tid >> 4, kcol = (tid & 15) * 8;         // sK staging coords
  const int vrow = tid >> 3, vcol = (tid & 7) * 8;          // sV staging coords

  for (int kt = 0; kt < SKL / 64; ++kt) {
    __syncthreads();                                        // prior iteration reads done
#if HAVE_ASYNC_LDS
#pragma unroll
    for (int it = 0; it < 4; ++it) {
      int kr = krow + it * 16;
      async_cp16(&Kp[(size_t)(kt * 64 + kr) * DM + kcol], &sK[kr * 136 + kcol]);
      int vr = vrow + it * 32;
      async_cp16(&Vp[(size_t)vr * SKL + kt * 64 + vcol], &sV[vr * 72 + vcol]);
    }
    __builtin_amdgcn_s_wait_asynccnt(0);
#else
#pragma unroll
    for (int it = 0; it < 4; ++it) {
      int kr = krow + it * 16;
      *(int4*)&sK[kr * 136 + kcol] = *(const int4*)&Kp[(size_t)(kt * 64 + kr) * DM + kcol];
      int vr = vrow + it * 32;
      *(int4*)&sV[vr * 72 + vcol] = *(const int4*)&Vp[(size_t)vr * SKL + kt * 64 + vcol];
    }
    if (kt + 1 < SKL / 64) {                                // warm L2/WGP$ for next tile
      __builtin_prefetch(&Kp[(size_t)((kt + 1) * 64 + krow) * DM + kcol], 0, 1);
      __builtin_prefetch(&Vp[(size_t)vrow * SKL + (kt + 1) * 64 + vcol], 0, 1);
    }
#endif
    __syncthreads();

    fx8 sacc[4];
#pragma unroll
    for (int ns = 0; ns < 4; ++ns) sacc[ns] = {};
#pragma unroll
    for (int ks = 0; ks < 4; ++ks)
#pragma unroll
      for (int ns = 0; ns < 4; ++ns) {
        bfx16 kf = ld_bfrag((const __bf16*)&sK[(ns * 16 + (lane & 15)) * 136 + ks * 32], lane);
        sacc[ns] = wmma_bf16(qf[ks], kf, sacc[ns]);
      }

    // online softmax (width-16 row reductions within wave32 halves)
    float mnew[8], rs[8];
#pragma unroll
    for (int r = 0; r < 8; ++r) {
      float t = fmaxf(fmaxf(sacc[0][r], sacc[1][r]), fmaxf(sacc[2][r], sacc[3][r])) * scale;
#pragma unroll
      for (int off = 1; off < 16; off <<= 1) t = fmaxf(t, __shfl_xor(t, off, 16));
      mnew[r] = fmaxf(m[r], t);
      rs[r] = 0.f;
    }
    const int mhi = (lane >= 16) ? 8 : 0;
#pragma unroll
    for (int ns = 0; ns < 4; ++ns)
#pragma unroll
      for (int r = 0; r < 8; ++r) {
        float p = __expf(sacc[ns][r] * scale - mnew[r]);
        rs[r] += p;
        myP[(r + mhi) * 72 + ns * 16 + (lane & 15)] = f2bf(p);
      }
#pragma unroll
    for (int r = 0; r < 8; ++r) {
      float t = rs[r];
#pragma unroll
      for (int off = 1; off < 16; off <<= 1) t += __shfl_xor(t, off, 16);
      float corr = __expf(m[r] - mnew[r]);
      l[r] = l[r] * corr + t;
      m[r] = mnew[r];
#pragma unroll
      for (int j = 0; j < 8; ++j) cacc[j][r] *= corr;
    }

    asm volatile("s_wait_dscnt 0" ::: "memory");            // wave-local LDS RAW on sP
    bfx16 pf[2];
#pragma unroll
    for (int ks = 0; ks < 2; ++ks)
      pf[ks] = ld_afrag((const __bf16*)&myP[(lane & 15) * 72 + ks * 32], lane);
#pragma unroll
    for (int ks = 0; ks < 2; ++ks)
#pragma unroll
      for (int ns = 0; ns < 8; ++ns) {
        bfx16 vf = ld_bfrag((const __bf16*)&sV[(ns * 16 + (lane & 15)) * 72 + ks * 32], lane);
        cacc[ns] = wmma_bf16(pf[ks], vf, cacc[ns]);
      }
  }

  unsigned short* Cp = CTX + (hb * SQL + q0) * DM;
  const int mhi = (lane >= 16) ? 8 : 0;
#pragma unroll
  for (int ns = 0; ns < 8; ++ns) {
    int d = ns * 16 + (lane & 15);
#pragma unroll
    for (int r = 0; r < 8; ++r)
      Cp[(size_t)(w * 16 + r + mhi) * DM + d] = f2bf(cacc[ns][r] / l[r]);
  }
}

// ---------- host ----------
extern "C" void kernel_launch(void* const* d_in, const int* in_sizes, int n_in,
                              void* d_out, int out_size, void* d_ws, size_t ws_size,
                              hipStream_t stream) {
  const float* query = (const float*)d_in[0];
  const float* key   = (const float*)d_in[1];
  const float* Wq = (const float*)d_in[2];  const float* bq = (const float*)d_in[3];
  const float* Wk = (const float*)d_in[4];  const float* bk = (const float*)d_in[5];
  const float* Wv = (const float*)d_in[6];  const float* bv = (const float*)d_in[7];
  const float* Wp = (const float*)d_in[8];  const float* bp = (const float*)d_in[9];
  const float* Wo = (const float*)d_in[10]; const float* bo = (const float*)d_in[11];
  float* out = (float*)d_out;

  char* ws = (char*)d_ws;
  const size_t MB = 1ull << 20;            // total workspace use: ~205.5 MB
  unsigned short* qA   = (unsigned short*)(ws + 0);        // [8192][1024] bf16
  unsigned short* kA   = (unsigned short*)(ws + 16 * MB);  // [8192][1024]
  unsigned short* Qb   = (unsigned short*)(ws + 32 * MB);  // [H][B][SQ][128]
  unsigned short* Kb   = (unsigned short*)(ws + 64 * MB);  // [H][B][SK][128]
  unsigned short* Vt   = (unsigned short*)(ws + 96 * MB);  // [H][B][128][SK]
  unsigned short* CTX  = (unsigned short*)(ws + 128 * MB); // [H][B][SQ][128]
  unsigned short* CAT  = (unsigned short*)(ws + 160 * MB); // [B*SQ][2048]
  unsigned short* WqvT = (unsigned short*)(ws + 192 * MB); // [4096][1024]
  unsigned short* WkT  = (unsigned short*)(ws + 200 * MB); // [2048][1024]
  unsigned short* WpT  = (unsigned short*)(ws + 204 * MB); // [16][128][128]
  unsigned short* WoT  = (unsigned short*)(ws + 205 * MB); // [128][2048]

  const size_t GEMM_LDS = 4 * 18432;       // double-buffered sA/sB (73728 B)

  k_cast<<<8192, 256, 0, stream>>>(query, qA, BB * SQL * INDIM);
  k_cast<<<8192, 256, 0, stream>>>(key,   kA, BB * SKL * INDIM);
  k_pack_wqv<<<16384, 256, 0, stream>>>(Wq, Wv, WqvT);
  k_pack_wk<<<8192, 256, 0, stream>>>(Wk, WkT);
  k_pack_wp<<<1024, 256, 0, stream>>>(Wp, WpT);
  k_pack_wo<<<1024, 256, 0, stream>>>(Wo, WoT);

  // Q|V projection: [8192 x 1024] x [1024 x 4096]
  k_gemm<<<dim3(32, 64), 256, GEMM_LDS, stream>>>(qA, INDIM, WqvT, INDIM, INDIM, 0,
                                                  bq, bv, Qb, Vt, nullptr);
  // K projection: [8192 x 1024] x [1024 x 2048]
  k_gemm<<<dim3(16, 64), 256, GEMM_LDS, stream>>>(kA, INDIM, WkT, INDIM, INDIM, 1,
                                                  bk, nullptr, Kb, nullptr, nullptr);
  // attention
  k_attn<<<dim3(SQL / 128, BB, NH), 256, 0, stream>>>(Qb, Kb, Vt, CTX);
  // per-head Wp -> concat: [131072 x 128] x [128 x 128]
  k_gemm<<<dim3(1, 1024), 256, GEMM_LDS, stream>>>(CTX, DM, WpT, DM, DM, 2,
                                                   bp, nullptr, CAT, nullptr, nullptr);
  // output projection: [8192 x 2048] x [2048 x 128] -> fp32 out
  k_gemm<<<dim3(1, 64), 256, GEMM_LDS, stream>>>(CAT, 2048, WoT, 2048, 2048, 3,
                                                 bo, nullptr, nullptr, nullptr, out);
}
